// mHCBlock_46007689674774
// MI455X (gfx1250) — matrix operand order repaired and checked by reading, not compiled, not run on hardware
//
#include <hip/hip_runtime.h>
#include <hip/hip_bf16.h>
#include <math.h>

// ---------------------------------------------------------------------------
// mHC block for MI455X (gfx1250, wave32, WMMA).
// GEMMs run as bf16 WMMA (v_wmma_f32_16x16x32_bf16) with f32 accumulation;
// everything else (LN, Sinkhorn, softmax, stream mixing) stays f32.
// ---------------------------------------------------------------------------

typedef __bf16 bf16;
typedef __attribute__((ext_vector_type(8)))  __bf16 v8bf;
typedef __attribute__((ext_vector_type(16))) __bf16 v16bf;
typedef __attribute__((ext_vector_type(8)))  float  v8f;

#define DMODEL 2048
#define NSTREAM 4
#define SDIM 512
#define MROWS 16384            // B*N = 4*4096

// workspace layout (bytes)
#define WS_SMAT   ((size_t)0)                               // 128 floats of tiny matrices
#define WS_WT_A   ((size_t)4096)                            // W_attn^T bf16 (N x K)
#define WS_WT_H   (WS_WT_A + (size_t)DMODEL*DMODEL*2)
#define WS_A_A    (WS_WT_H + (size_t)DMODEL*DMODEL*2)       // attn_in  bf16 (M x K)
#define WS_A_H    (WS_A_A  + (size_t)MROWS*DMODEL*2)        // hyb_in   bf16
#define WS_Y_A    (WS_A_H  + (size_t)MROWS*DMODEL*2)        // attn_out bf16
#define WS_Y_H    (WS_Y_A  + (size_t)MROWS*DMODEL*2)        // hyb_out  bf16

// smat float offsets
#define SM_PRE_A  0
#define SM_PRE_H  16
#define SM_RES_A  32
#define SM_RES_H  48
#define SM_POST_A 64
#define SM_POST_H 80
#define SM_ALPHA  96

// ---------------------------------------------------------------------------
// Kernel 0: all 4x4 matrix math (Sinkhorn 20 iters, row-softmaxes, sigmoid).
// Negligible work -> single lane.
// ---------------------------------------------------------------------------
__global__ void mhc_small_mats_kernel(const float* __restrict__ pre_a_l,
                                      const float* __restrict__ pre_h_l,
                                      const float* __restrict__ res_a_l,
                                      const float* __restrict__ res_h_l,
                                      const float* __restrict__ post_a_l,
                                      const float* __restrict__ post_h_l,
                                      const float* __restrict__ cross_mix,
                                      float* __restrict__ smat) {
  if (threadIdx.x != 0) return;

  auto softmax_rows = [](const float* L, float* O) {
    for (int r = 0; r < 4; ++r) {
      float mx = L[r*4];
      for (int c = 1; c < 4; ++c) mx = fmaxf(mx, L[r*4+c]);
      float e[4]; float s = 0.f;
      for (int c = 0; c < 4; ++c) { e[c] = __expf(L[r*4+c] - mx); s += e[c]; }
      float inv = 1.f / s;
      for (int c = 0; c < 4; ++c) O[r*4+c] = e[c] * inv;
    }
  };
  auto sinkhorn = [](const float* L, float* O) {
    float mx = L[0];
    for (int i = 1; i < 16; ++i) mx = fmaxf(mx, L[i]);
    float P[16];
    for (int i = 0; i < 16; ++i) P[i] = __expf(L[i] - mx);
    for (int it = 0; it < 20; ++it) {
      for (int r = 0; r < 4; ++r) {
        float s = 1e-8f;
        for (int c = 0; c < 4; ++c) s += P[r*4+c];
        float inv = 1.f / s;
        for (int c = 0; c < 4; ++c) P[r*4+c] *= inv;
      }
      for (int c = 0; c < 4; ++c) {
        float s = 1e-8f;
        for (int r = 0; r < 4; ++r) s += P[r*4+c];
        float inv = 1.f / s;
        for (int r = 0; r < 4; ++r) P[r*4+c] *= inv;
      }
    }
    for (int i = 0; i < 16; ++i) O[i] = P[i];
  };

  softmax_rows(pre_a_l,  smat + SM_PRE_A);
  softmax_rows(pre_h_l,  smat + SM_PRE_H);
  sinkhorn    (res_a_l,  smat + SM_RES_A);
  sinkhorn    (res_h_l,  smat + SM_RES_H);
  softmax_rows(post_a_l, smat + SM_POST_A);
  softmax_rows(post_h_l, smat + SM_POST_H);
  smat[SM_ALPHA] = 1.f / (1.f + __expf(-cross_mix[0]));
}

// ---------------------------------------------------------------------------
// Kernel 1: W (K x N, f32) -> W^T (N x K, bf16), tiled through LDS so both
// global read and write stay coalesced. W^T makes WMMA B-fragments contiguous.
// ---------------------------------------------------------------------------
__global__ void mhc_wtrans_kernel(const float* __restrict__ W, bf16* __restrict__ Wt) {
  __shared__ float tile[32][33];
  const int bx = blockIdx.x * 32;   // n
  const int by = blockIdx.y * 32;   // k
  const int tx = threadIdx.x;       // 0..31
  const int ty = threadIdx.y;       // 0..7
  for (int i = 0; i < 32; i += 8)
    tile[ty + i][tx] = W[(size_t)(by + ty + i) * DMODEL + (bx + tx)];
  __syncthreads();
  for (int i = 0; i < 32; i += 8)
    Wt[(size_t)(bx + ty + i) * DMODEL + (by + tx)] = (bf16)tile[tx][ty + i];
}

// ---------------------------------------------------------------------------
// Kernel 2: fused LayerNorm + H_pre stream mixing -> bf16 GEMM inputs.
// One 256-thread block per token row (D=2048, 8 elements/thread).
//   attn_in[t*512+d] = sum_s LN(x)[s*512+d] * Hpre_a[s][t]
//   hyb_in [t*512+d] = sum_s      x[s*512+d] * Hpre_h[s][t]
// ---------------------------------------------------------------------------
__global__ void __launch_bounds__(256) mhc_ln_mix_kernel(
    const float* __restrict__ x, const float* __restrict__ gamma,
    const float* __restrict__ beta, const float* __restrict__ smat,
    bf16* __restrict__ Aa, bf16* __restrict__ Ah) {
  __shared__ float xr[DMODEL];
  __shared__ float xn[DMODEL];
  __shared__ float redS[256];
  __shared__ float redQ[256];

  const int row = blockIdx.x;
  const int tid = threadIdx.x;
  const float* xp = x + (size_t)row * DMODEL;

  float v[8];
  float s = 0.f, sq = 0.f;
  for (int i = 0; i < 8; ++i) {
    int j = i * 256 + tid;
    float f = xp[j];
    v[i] = f; xr[j] = f;
    s += f; sq += f * f;
  }
  redS[tid] = s; redQ[tid] = sq;
  __syncthreads();
  for (int off = 128; off > 0; off >>= 1) {
    if (tid < off) { redS[tid] += redS[tid + off]; redQ[tid] += redQ[tid + off]; }
    __syncthreads();
  }
  const float mu  = redS[0] * (1.f / DMODEL);
  const float var = redQ[0] * (1.f / DMODEL) - mu * mu;
  const float rs  = rsqrtf(var + 1e-5f);

  for (int i = 0; i < 8; ++i) {
    int j = i * 256 + tid;
    xn[j] = (v[i] - mu) * rs * gamma[j] + beta[j];
  }
  __syncthreads();

  float pa[16], ph[16];
  for (int i = 0; i < 16; ++i) { pa[i] = smat[SM_PRE_A + i]; ph[i] = smat[SM_PRE_H + i]; }

  bf16* Aar = Aa + (size_t)row * DMODEL;
  bf16* Ahr = Ah + (size_t)row * DMODEL;
  for (int i = 0; i < 8; ++i) {
    int j = i * 256 + tid;
    int t = j >> 9;
    int d = j & (SDIM - 1);
    float aa = 0.f, ah = 0.f;
    for (int ss = 0; ss < NSTREAM; ++ss) {
      aa += xn[ss * SDIM + d] * pa[ss * 4 + t];
      ah += xr[ss * SDIM + d] * ph[ss * 4 + t];
    }
    Aar[j] = (bf16)aa;
    Ahr[j] = (bf16)ah;
  }
}

// ---------------------------------------------------------------------------
// Kernel 3: Y = A(bf16, MxK) * W^T(bf16, NxK)^T + bias, via WMMA bf16.
// Block = 8 waves; each wave owns a 64x64 tile (4x4 of 16x16x32 WMMAs).
// Fragment layouts used here (contiguous K, same order in A and B):
//   A 16x32: lanes 0-15 -> row m, K{0..15};       lanes 16-31 -> K{16..31}
//   B 32x16: lanes 0-15 -> col n, K{0..15};       lanes 16-31 -> K{16..31}
// Both GEMMs (attn / hyb) share this kernel via blockIdx.z.
// ---------------------------------------------------------------------------
__global__ void __launch_bounds__(256) mhc_gemm_wmma_kernel(
    const bf16* __restrict__ Aa, const bf16* __restrict__ Ah,
    const bf16* __restrict__ Wta, const bf16* __restrict__ Wth,
    const float* __restrict__ ba, const float* __restrict__ bh,
    bf16* __restrict__ Ya, bf16* __restrict__ Yh) {
  const int path = blockIdx.z;
  const bf16* A    = path ? Ah  : Aa;
  const bf16* Wt   = path ? Wth : Wta;
  const float* bias = path ? bh : ba;
  bf16* Y          = path ? Yh  : Ya;

  const int lane  = threadIdx.x & 31;
  const int wave  = threadIdx.x >> 5;
  const int m0    = blockIdx.y * 64;
  const int n0    = blockIdx.x * 512 + wave * 64;
  const int lhalf = lane >> 4;
  const int lrow  = lane & 15;
  const int akoff = lhalf ? 16 : 0;   // A fragment K offset for upper lanes
  const int bkoff = lhalf ? 16 : 0;   // B fragment K offset for upper lanes

  const bf16* arow[4];
  const bf16* brow[4];
  for (int mi = 0; mi < 4; ++mi) arow[mi] = A  + (size_t)(m0 + mi * 16 + lrow) * DMODEL;
  for (int ni = 0; ni < 4; ++ni) brow[ni] = Wt + (size_t)(n0 + ni * 16 + lrow) * DMODEL;

  v8f acc[4][4];
  const v8f vzero = {0.f, 0.f, 0.f, 0.f, 0.f, 0.f, 0.f, 0.f};
  for (int mi = 0; mi < 4; ++mi)
    for (int ni = 0; ni < 4; ++ni)
      acc[mi][ni] = vzero;

  for (int k0 = 0; k0 < DMODEL; k0 += 32) {
    // speculative L2 prefetch of the next K tile (global_prefetch_b8)
    __builtin_prefetch((const void*)(arow[0] + k0 + 32 + akoff), 0, 0);
    __builtin_prefetch((const void*)(brow[0] + k0 + 32 + bkoff), 0, 0);

    v16bf afrag[4], bfrag[4];
    for (int mi = 0; mi < 4; ++mi) {
      v8bf lo = *(const v8bf*)(arow[mi] + k0 + akoff);        // K = koff..koff+7
      v8bf hi = *(const v8bf*)(arow[mi] + k0 + akoff + 8);    // K = koff+8..koff+15
      afrag[mi] = __builtin_shufflevector(lo, hi, 0, 1, 2, 3, 4, 5, 6, 7,
                                                  8, 9, 10, 11, 12, 13, 14, 15);
    }
    for (int ni = 0; ni < 4; ++ni) {
      bfrag[ni] = *(const v16bf*)(brow[ni] + k0 + bkoff);     // 16 contiguous K
    }
    for (int mi = 0; mi < 4; ++mi)
      for (int ni = 0; ni < 4; ++ni)
        acc[mi][ni] = __builtin_amdgcn_wmma_f32_16x16x32_bf16(
            false, afrag[mi], false, bfrag[ni], (short)0, acc[mi][ni], false, false);
  }

  // Fused bias + bf16 store. C/D layout: VGPR r -> M = r (+8 for lanes 16-31),
  // N = lane & 15.
  for (int ni = 0; ni < 4; ++ni) {
    const int col = n0 + ni * 16 + lrow;
    const float bv = bias[col];
    for (int mi = 0; mi < 4; ++mi) {
      for (int r = 0; r < 8; ++r) {
        const int rowm = m0 + mi * 16 + r + (lhalf ? 8 : 0);
        Y[(size_t)rowm * DMODEL + col] = (bf16)(acc[mi][ni][r] + bv);
      }
    }
  }
}

// ---------------------------------------------------------------------------
// Kernel 4: epilogue.  out[t,d] = alpha * (sum_s x[s,d] Hres_a[s][t]
//                                         + scl_a[t] sum_s Ya[s,d] Hpost_a[t][s])
//                             + (1-alpha) * (same with hyb)
// One thread per (row, d); produces all 4 stream outputs.
// ---------------------------------------------------------------------------
__global__ void __launch_bounds__(256) mhc_epilogue_kernel(
    const float* __restrict__ x, const bf16* __restrict__ Ya,
    const bf16* __restrict__ Yh, const float* __restrict__ smat,
    const float* __restrict__ scl_a, const float* __restrict__ scl_h,
    float* __restrict__ out) {
  const int gid = blockIdx.x * 256 + threadIdx.x;
  const int row = gid >> 9;
  const int d   = gid & (SDIM - 1);

  const float* xr  = x  + (size_t)row * DMODEL;
  const bf16*  yar = Ya + (size_t)row * DMODEL;
  const bf16*  yhr = Yh + (size_t)row * DMODEL;

  float xs[4], yas[4], yhs[4];
  for (int ss = 0; ss < NSTREAM; ++ss) {
    const int j = ss * SDIM + d;
    xs[ss]  = xr[j];
    yas[ss] = (float)yar[j];
    yhs[ss] = (float)yhr[j];
  }
  const float alpha = smat[SM_ALPHA];
  float* outr = out + (size_t)row * DMODEL;
  for (int t = 0; t < NSTREAM; ++t) {
    float ra = 0.f, rh = 0.f, ma = 0.f, mh = 0.f;
    for (int ss = 0; ss < NSTREAM; ++ss) {
      ra += xs[ss]  * smat[SM_RES_A  + ss * 4 + t];
      rh += xs[ss]  * smat[SM_RES_H  + ss * 4 + t];
      ma += yas[ss] * smat[SM_POST_A + t * 4 + ss];
      mh += yhs[ss] * smat[SM_POST_H + t * 4 + ss];
    }
    const float va = ra + scl_a[t] * ma;
    const float vh = rh + scl_h[t] * mh;
    outr[t * SDIM + d] = alpha * va + (1.f - alpha) * vh;
  }
}

// ---------------------------------------------------------------------------
extern "C" void kernel_launch(void* const* d_in, const int* in_sizes, int n_in,
                              void* d_out, int out_size, void* d_ws, size_t ws_size,
                              hipStream_t stream) {
  (void)in_sizes; (void)n_in; (void)out_size; (void)ws_size;

  const float* x      = (const float*)d_in[0];
  const float* gamma  = (const float*)d_in[1];
  const float* beta   = (const float*)d_in[2];
  const float* W_attn = (const float*)d_in[3];
  const float* b_attn = (const float*)d_in[4];
  const float* W_hyb  = (const float*)d_in[5];
  const float* b_hyb  = (const float*)d_in[6];
  const float* pre_a  = (const float*)d_in[7];
  const float* pre_h  = (const float*)d_in[8];
  const float* res_a  = (const float*)d_in[9];
  const float* post_a = (const float*)d_in[10];
  const float* scl_a  = (const float*)d_in[11];
  const float* res_h  = (const float*)d_in[12];
  const float* post_h = (const float*)d_in[13];
  const float* scl_h  = (const float*)d_in[14];
  const float* cmix   = (const float*)d_in[15];

  char* ws = (char*)d_ws;
  float* smat = (float*)(ws + WS_SMAT);
  bf16*  Wta  = (bf16*)(ws + WS_WT_A);
  bf16*  Wth  = (bf16*)(ws + WS_WT_H);
  bf16*  Aa   = (bf16*)(ws + WS_A_A);
  bf16*  Ah   = (bf16*)(ws + WS_A_H);
  bf16*  Ya   = (bf16*)(ws + WS_Y_A);
  bf16*  Yh   = (bf16*)(ws + WS_Y_H);
  float* out  = (float*)d_out;

  // 0) tiny 4x4 math
  mhc_small_mats_kernel<<<1, 32, 0, stream>>>(pre_a, pre_h, res_a, res_h,
                                              post_a, post_h, cmix, smat);
  // 1) weight transpose + bf16 convert
  {
    dim3 grid(DMODEL / 32, DMODEL / 32);
    dim3 block(32, 8);
    mhc_wtrans_kernel<<<grid, block, 0, stream>>>(W_attn, Wta);
    mhc_wtrans_kernel<<<grid, block, 0, stream>>>(W_hyb,  Wth);
  }
  // 2) fused LayerNorm + H_pre mixing -> bf16 GEMM inputs
  mhc_ln_mix_kernel<<<MROWS, 256, 0, stream>>>(x, gamma, beta, smat, Aa, Ah);
  // 3) both GEMMs via WMMA (z = path)
  {
    dim3 grid(DMODEL / 512, MROWS / 64, 2);
    mhc_gemm_wmma_kernel<<<grid, 256, 0, stream>>>(Aa, Ah, Wta, Wth,
                                                   b_attn, b_hyb, Ya, Yh);
  }
  // 4) residual + post-mixing + alpha blend
  mhc_epilogue_kernel<<<(MROWS * SDIM) / 256, 256, 0, stream>>>(
      x, Ya, Yh, smat, scl_a, scl_h, out);
}
